// MatchingLayer_33122787787582
// MI455X (gfx1250) — matrix operand order repaired and checked by reading, not compiled, not run on hardware
//
#include <hip/hip_runtime.h>
#include <hip/hip_bf16.h>

// ---------------------------------------------------------------------------
// MatchingLayer on MI455X (gfx1250):
//   sim = normalize(s) @ normalize(q)^T  (9216x9216, K=256) fused with
//   per-row top-20 mean over fg and bg columns.
// v_wmma_f32_16x16x32_f16; 32 rows/block (2 A tiles in VGPRs) to halve L2
// traffic; B tiles ping-pong double-buffered across column chunks so global
// load latency hides under the scan/barrier phase.
// ---------------------------------------------------------------------------

typedef _Float16 half8  __attribute__((ext_vector_type(8)));
typedef _Float16 v16h_t __attribute__((ext_vector_type(16)));
typedef float    v8f_t  __attribute__((ext_vector_type(8)));

union V16U { half8 h[2]; v16h_t v; };

#define MPIX 9216      // 96*96
#define CDIM 256
#define NEGV (-1e30f)

// ---------------------------------------------------------------- mask ------
__global__ __launch_bounds__(256) void mask_kernel(
    const int* __restrict__ ql, const int* __restrict__ color,
    int* __restrict__ mask, int M) {
  int m = blockIdx.x * blockDim.x + threadIdx.x;
  if (m >= M) return;
  int c0 = color[0], c1 = color[1], c2 = color[2];
  mask[m] = (ql[3 * m + 0] == c0 && ql[3 * m + 1] == c1 && ql[3 * m + 2] == c2)
                ? 1 : 0;
}

// ------------------------------------------------------- normalize rows -----
// feat is [C, M] (channel-major, from [1,C,h,w]); out is [M, C] row-major f16.
__global__ __launch_bounds__(256) void norm_rows_kernel(
    const float* __restrict__ feat, _Float16* __restrict__ out, int M, int C) {
  int m = blockIdx.x * blockDim.x + threadIdx.x;
  if (m >= M) return;
  float s = 0.f;
  for (int c = 0; c < C; ++c) {
    float v = feat[(size_t)c * M + m];   // coalesced across lanes
    s += v * v;
  }
  float n = sqrtf(s);
  float scale = 1.0f / fmaxf(n, 1e-12f);
  for (int c = 0; c < C; ++c) {
    out[(size_t)m * C + c] = (_Float16)(feat[(size_t)c * M + m] * scale);
  }
}

// --------------------------------------------------------------- helpers ----
// Load one wave's 16-column B tile for a column base: 8 K-step fragments.
__device__ __forceinline__ void load_btile(V16U (&B)[8],
                                           const _Float16* __restrict__ qn,
                                           int colbase, int wave, int rsel,
                                           int hi) {
  const int mc = colbase + wave * 16 + rsel;
  const _Float16* br = qn + (size_t)mc * CDIM + hi * 8;
#pragma unroll
  for (int k = 0; k < 8; ++k) {
    B[k].h[0] = *(const half8*)(br + 32 * k);
    B[k].h[1] = *(const half8*)(br + 32 * k + 16);
  }
}

// Compute two 16x16 WMMA tiles against one B tile, mask into LDS, run the
// per-list top-20 scan. 64 lists (32 rows x fg/bg) x 4 sub-threads.
__device__ __forceinline__ void compute_scan(
    const V16U (&A0)[8], const V16U (&A1)[8], const V16U (&B)[8],
    int m0, const int* __restrict__ mask,
    float* sFG, float* sBG, float* sTop,
    float& minVal, int& minIdx,
    int tid, int wave, int rsel, int hi) {
  const int mcol = m0 + wave * 16 + rsel;
  const int mk = mask[mcol];           // issued early, overlaps WMMAs

  v8f_t acc0 = {0.f, 0.f, 0.f, 0.f, 0.f, 0.f, 0.f, 0.f};
  v8f_t acc1 = {0.f, 0.f, 0.f, 0.f, 0.f, 0.f, 0.f, 0.f};
#pragma unroll
  for (int k = 0; k < 8; ++k) {
    acc0 = __builtin_amdgcn_wmma_f32_16x16x32_f16(
        false, A0[k].v, false, B[k].v, (short)0, acc0, false, false);
    acc1 = __builtin_amdgcn_wmma_f32_16x16x32_f16(
        false, A1[k].v, false, B[k].v, (short)0, acc1, false, false);
  }

  const int col = wave * 16 + rsel;
#pragma unroll
  for (int j = 0; j < 8; ++j) {
    const int row0 = j + hi * 8;       // rows 0..15  (tile 0)
    const int row1 = 16 + row0;        // rows 16..31 (tile 1)
    float v0 = acc0[j];
    float v1 = acc1[j];
    sFG[row0 * 128 + col] = mk ? v0 : NEGV;
    sBG[row0 * 128 + col] = mk ? NEGV : v0;
    sFG[row1 * 128 + col] = mk ? v1 : NEGV;
    sBG[row1 * 128 + col] = mk ? NEGV : v1;
  }
  __syncthreads();

  {
    const int list = tid >> 2;         // 0..63
    const int sub  = tid & 3;
    const int row  = list & 31;
    const float* src = (list < 32) ? (sFG + row * 128) : (sBG + row * 128);
    const int cbase = sub * 32;
#pragma unroll
    for (int c = 0; c < 32; ++c) {
      float v = src[cbase + c];
      if (v > minVal) {
        sTop[tid * 20 + minIdx] = v;
        float mv = 3e38f; int mi = 0;
#pragma unroll
        for (int i = 0; i < 20; ++i) {
          float t = sTop[tid * 20 + i];
          if (t < mv) { mv = t; mi = i; }
        }
        minVal = mv; minIdx = mi;
      }
    }
  }
  __syncthreads();
}

// ------------------------------------------- fused GEMM + masked top-K ------
__global__ __launch_bounds__(256) void gemm_topk_kernel(
    const _Float16* __restrict__ sn,  // [N, C] rows of sim (A)
    const _Float16* __restrict__ qn,  // [M, C] cols of sim (B)
    const int* __restrict__ mask,     // [M]
    float* __restrict__ out,          // [2*N]: fg then bg
    int N) {
  __shared__ float sFG[32 * 128];     // 16 KB
  __shared__ float sBG[32 * 128];     // 16 KB
  __shared__ float sTop[256 * 20];    // 20 KB

  const int tid  = threadIdx.x;
  const int lane = tid & 31;
  const int wave = tid >> 5;
  const int n0   = blockIdx.x * 32;
  const int rsel = lane & 15;
  const int hi   = lane >> 4;

#pragma unroll
  for (int i = 0; i < 20; ++i) sTop[tid * 20 + i] = NEGV;
  float minVal = NEGV;
  int   minIdx = 0;

  // Two A tiles (rows n0..n0+15 and n0+16..n0+31) resident in VGPRs.
  // A lane layout: lanes 0-15 K={k..k+7, k+16..k+23}; lanes 16-31 +8.
  const _Float16* aRow0 = sn + (size_t)(n0 + rsel) * CDIM + hi * 8;
  const _Float16* aRow1 = aRow0 + 16 * CDIM;
  V16U A0[8], A1[8];
#pragma unroll
  for (int k = 0; k < 8; ++k) {
    A0[k].h[0] = *(const half8*)(aRow0 + 32 * k);
    A0[k].h[1] = *(const half8*)(aRow0 + 32 * k + 16);
    A1[k].h[0] = *(const half8*)(aRow1 + 32 * k);
    A1[k].h[1] = *(const half8*)(aRow1 + 32 * k + 16);
  }

  // Ping-pong B tiles: loads for the next chunk are issued before the
  // current chunk's scan/barrier phase, hiding global latency.
  V16U Bc[8], Bn[8];
  load_btile(Bc, qn, 0, wave, rsel, hi);

  for (int m0 = 0; m0 < N; m0 += 256) {
    const int p1 = m0 + 128;                       // always < N (N % 256 == 0)
    load_btile(Bn, qn, p1, wave, rsel, hi);
    compute_scan(A0, A1, Bc, m0, mask, sFG, sBG, sTop,
                 minVal, minIdx, tid, wave, rsel, hi);

    int p2 = m0 + 256;
    if (p2 >= N) p2 = 0;                           // harmless wrap prefetch
    load_btile(Bc, qn, p2, wave, rsel, hi);
    compute_scan(A0, A1, Bn, p1, mask, sFG, sBG, sTop,
                 minVal, minIdx, tid, wave, rsel, hi);
  }

  // Merge: each list's 4 partial top-20 lists are contiguous (80 floats).
  if (tid < 64) {
    const int list = tid;
    const int base = list * 80;
    float accum = 0.f;
    for (int r = 0; r < 20; ++r) {
      float best = -3e38f; int bi = 0;
      for (int i = 0; i < 80; ++i) {
        float v = sTop[base + i];
        if (v > best) { best = v; bi = i; }
      }
      accum += best;
      sTop[base + bi] = -3e38f;
    }
    const int row = list & 31;
    out[((list < 32) ? 0 : N) + n0 + row] = accum * (1.0f / 20.0f);
  }
}

// ---------------------------------------------------------------------------
extern "C" void kernel_launch(void* const* d_in, const int* in_sizes, int n_in,
                              void* d_out, int out_size, void* d_ws, size_t ws_size,
                              hipStream_t stream) {
  (void)in_sizes; (void)n_in; (void)out_size; (void)ws_size;
  const int*   query_label = (const int*)d_in[0];   // [96*96*3]
  const int*   color       = (const int*)d_in[1];   // [3]
  const float* q_feat      = (const float*)d_in[2]; // [1,256,96,96]
  const float* s_feat      = (const float*)d_in[3]; // [1,256,96,96]
  float*       out         = (float*)d_out;         // [2*9216]

  // workspace layout: qn f16 [M,C], sn f16 [M,C], mask int [M]
  _Float16* qn   = (_Float16*)d_ws;
  _Float16* sn   = qn + (size_t)MPIX * CDIM;
  int*      mask = (int*)(sn + (size_t)MPIX * CDIM);

  const int threads = 256;
  const int mblocks = (MPIX + threads - 1) / threads;  // 36

  mask_kernel<<<mblocks, threads, 0, stream>>>(query_label, color, mask, MPIX);
  norm_rows_kernel<<<mblocks, threads, 0, stream>>>(q_feat, qn, MPIX, CDIM);
  norm_rows_kernel<<<mblocks, threads, 0, stream>>>(s_feat, sn, MPIX, CDIM);

  // 9216 rows / 32 rows-per-block = 288 blocks
  gemm_topk_kernel<<<MPIX / 32, threads, 0, stream>>>(sn, qn, mask, out, MPIX);
}
